// PonderRelationalGraphConvModel_36988258353759
// MI455X (gfx1250) — compile-verified, matrix operand order checked by reading, not run on hardware
//
#include <hip/hip_runtime.h>
#include <math.h>

#define NN 100000          // N_NODES
#define NE 250000          // N_EDGES
#define EMB 32
#define NRELS 200          // NUM_RELS2
#define NTYPES 16
#define SPLIT 8            // grid split per relation

typedef _Float16 v16h __attribute__((ext_vector_type(16)));
typedef _Float16 v8h  __attribute__((ext_vector_type(8)));
typedef float    v8f  __attribute__((ext_vector_type(8)));

// ---------------- gather h0 = entity[src_ids], f32 -> f16 ----------------
__global__ void k_gather_h0(const float* __restrict__ entity,
                            const int* __restrict__ src_ids,
                            _Float16* __restrict__ h0) {
  int i = blockIdx.x * blockDim.x + threadIdx.x;
  if (i >= NN * EMB) return;
  int n = i >> 5, c = i & 31;
  h0[i] = (_Float16)entity[(size_t)src_ids[n] * EMB + c];
}

// ---------------- bucket sort edges by etype ----------------
__global__ void k_hist(const int* __restrict__ etype, int* __restrict__ hist) {
  int e = blockIdx.x * blockDim.x + threadIdx.x;
  if (e >= NE) return;
  atomicAdd(&hist[etype[e]], 1);
}

__global__ void k_scan(const int* __restrict__ hist, int* __restrict__ offs,
                       int* __restrict__ cursor) {
  if (blockIdx.x == 0 && threadIdx.x == 0) {
    int acc = 0;
    for (int r = 0; r < NRELS; ++r) {
      offs[r] = acc; cursor[r] = acc; acc += hist[r];
    }
    offs[NRELS] = acc;
  }
}

__global__ void k_scatter(const int* __restrict__ etype, int* __restrict__ cursor,
                          int* __restrict__ perm) {
  int e = blockIdx.x * blockDim.x + threadIdx.x;
  if (e >= NE) return;
  int pos = atomicAdd(&cursor[etype[e]], 1);
  perm[pos] = e;
}

// ---------------- degree (shared by both layers) ----------------
__global__ void k_deg(const int* __restrict__ edge_dst, float* __restrict__ deg) {
  int e = blockIdx.x * blockDim.x + threadIdx.x;
  if (e >= NE) return;
  atomicAdd(&deg[edge_dst[e]], 1.0f);
}

// ---------------- per-relation WMMA GEMM + atomic scatter-add ----------------
// One wave processes a 16-edge tile: A = 16x32 (f16 rows of h[src]),
// B = 32xOUT slice of W[r] (f16), D accumulated f32, scattered via atomics.
template <int OUT>
__global__ void k_rel_gemm(const _Float16* __restrict__ hsrc,
                           const float* __restrict__ W,
                           const int* __restrict__ perm,
                           const int* __restrict__ offs,
                           const int* __restrict__ edge_src,
                           const int* __restrict__ edge_dst,
                           float* __restrict__ sOut) {
  const int lane = threadIdx.x & 31;
  const int wave = threadIdx.x >> 5;           // 0..7 (256 threads)
  const int r    = blockIdx.x / SPLIT;
  const int part = blockIdx.x % SPLIT;
  const int wavesPerRel = SPLIT * 8;
  const int waveSlot    = part * 8 + wave;

  const int start = offs[r];
  const int end   = offs[r + 1];
  const int len   = end - start;
  if (len <= 0) return;                         // uniform per block
  const int ntiles = (len + 15) >> 4;

  const int g    = lane >> 4;                   // lane half-group (0/1)
  const int kg   = g << 3;                      // 0 or 8
  const int ncol = lane & 15;

  // Load B tile(s) of W[r] (32 x OUT) into WMMA B layout (mirror of A layout).
  const float* Wr = W + (size_t)r * EMB * OUT;
  v16h b0, b1;
  #pragma unroll
  for (int j = 0; j < 16; ++j) {
    int k = ((j < 8) ? 0 : 16) + kg + (j & 7);
    b0[j] = (_Float16)Wr[k * OUT + ncol];
    if (OUT == 32) b1[j] = (_Float16)Wr[k * OUT + 16 + ncol];
  }

  for (int t = waveSlot; t < ntiles; t += wavesPerRel) {
    // A: lanes L and L+16 own edge row m = L&15; each loads its k-half.
    int eidx = start + t * 16 + ncol;
    int e    = (eidx < end) ? perm[eidx] : -1;
    int dst  = -1;
    v16h a;
    #pragma unroll
    for (int j = 0; j < 16; ++j) a[j] = (_Float16)0.0f;
    if (e >= 0) {
      dst = edge_dst[e];
      const v8h* row = (const v8h*)(hsrc + (size_t)edge_src[e] * EMB);
      v8h lo = row[g];        // k = kg .. kg+7
      v8h hi = row[2 + g];    // k = 16+kg .. 16+kg+7
      #pragma unroll
      for (int j = 0; j < 8; ++j) { a[j] = lo[j]; a[j + 8] = hi[j]; }
    }

    v8f c0 = {};
    c0 = __builtin_amdgcn_wmma_f32_16x16x32_f16(false, a, false, b0,
                                                (short)0, c0, false, false);
    v8f c1 = {};
    if (OUT == 32)
      c1 = __builtin_amdgcn_wmma_f32_16x16x32_f16(false, a, false, b1,
                                                  (short)0, c1, false, false);

    // D layout: vgpr v holds row m = v + 8*g, column n = lane&15 (+16 for b1).
    #pragma unroll
    for (int v = 0; v < 8; ++v) {
      int m = v + (g << 3);
      int d = __shfl(dst, m, 32);               // dst of edge row m (lane m owns it)
      if (d >= 0) {
        atomicAdd(&sOut[(size_t)d * OUT + ncol], c0[v]);
        if (OUT == 32)
          atomicAdd(&sOut[(size_t)d * OUT + 16 + ncol], c1[v]);
      }
    }
  }
}

// ---------------- mean + relu -> f16 for layer-2 A ----------------
__global__ void k_mean_relu(const float* __restrict__ s1,
                            const float* __restrict__ deg,
                            _Float16* __restrict__ h1) {
  int i = blockIdx.x * blockDim.x + threadIdx.x;
  if (i >= NN * EMB) return;
  float v = s1[i] / fmaxf(deg[i >> 5], 1.0f);
  h1[i] = (_Float16)fmaxf(v, 0.0f);
}

// ---------------- layer-2 mean, write y[0] and y[1] ----------------
__global__ void k_mean_out(const float* __restrict__ s2,
                           const float* __restrict__ deg,
                           float* __restrict__ y) {
  int i = blockIdx.x * blockDim.x + threadIdx.x;
  if (i >= NN * NTYPES) return;
  float v = s2[i] / fmaxf(deg[i >> 4], 1.0f);
  y[i] = v;
  y[(size_t)NN * NTYPES + i] = v;
}

// ---------------- ponder head: p = [sigmoid, 1-sigmoid] ----------------
__global__ void k_ponder(const float* __restrict__ y,
                         const float* __restrict__ lw,
                         const float* __restrict__ lb,
                         float* __restrict__ p) {
  int n = blockIdx.x * blockDim.x + threadIdx.x;
  if (n >= NN) return;
  float acc = lb[0];
  #pragma unroll
  for (int c = 0; c < NTYPES; ++c) acc += y[(size_t)n * NTYPES + c] * lw[c];
  float lam = 1.0f / (1.0f + expf(-acc));
  p[n] = lam;
  p[NN + n] = 1.0f - lam;
}

extern "C" void kernel_launch(void* const* d_in, const int* in_sizes, int n_in,
                              void* d_out, int out_size, void* d_ws, size_t ws_size,
                              hipStream_t stream) {
  (void)in_sizes; (void)n_in; (void)out_size; (void)ws_size;
  const float* entity   = (const float*)d_in[0];   // [NN,32]
  const float* W1       = (const float*)d_in[1];   // [200,32,32]
  const float* W2       = (const float*)d_in[2];   // [200,32,16]
  const float* lw       = (const float*)d_in[3];   // [16,1]
  const float* lb       = (const float*)d_in[4];   // [1]
  const int*   src_ids  = (const int*)d_in[5];
  const int*   edge_src = (const int*)d_in[6];
  const int*   edge_dst = (const int*)d_in[7];
  const int*   etype    = (const int*)d_in[8];

  char* ws = (char*)d_ws;
  size_t off = 0;
  auto alloc = [&](size_t bytes) -> void* {
    off = (off + 255) & ~(size_t)255;
    void* p = ws + off;
    off += bytes;
    return p;
  };

  float*     s1     = (float*)alloc((size_t)NN * EMB * 4);
  float*     s2     = (float*)alloc((size_t)NN * NTYPES * 4);
  float*     deg    = (float*)alloc((size_t)NN * 4);
  _Float16*  h0     = (_Float16*)alloc((size_t)NN * EMB * 2);
  _Float16*  h1     = (_Float16*)alloc((size_t)NN * EMB * 2);
  int*       perm   = (int*)alloc((size_t)NE * 4);
  int*       hist   = (int*)alloc(NRELS * 4);
  int*       offs   = (int*)alloc((NRELS + 1) * 4);
  int*       cursor = (int*)alloc(NRELS * 4);

  float* y = (float*)d_out;                         // [2,NN,16]
  float* p = y + (size_t)2 * NN * NTYPES;           // [2,NN]

  const int TPB = 256;
  // gather input embeddings
  k_gather_h0<<<(NN * EMB + TPB - 1) / TPB, TPB, 0, stream>>>(entity, src_ids, h0);

  // bucket-sort edges by relation
  hipMemsetAsync(hist, 0, NRELS * 4, stream);
  k_hist<<<(NE + TPB - 1) / TPB, TPB, 0, stream>>>(etype, hist);
  k_scan<<<1, 1, 0, stream>>>(hist, offs, cursor);
  k_scatter<<<(NE + TPB - 1) / TPB, TPB, 0, stream>>>(etype, cursor, perm);

  // degrees
  hipMemsetAsync(deg, 0, (size_t)NN * 4, stream);
  k_deg<<<(NE + TPB - 1) / TPB, TPB, 0, stream>>>(edge_dst, deg);

  // layer 1: WMMA per-relation GEMM, atomic segment sum
  hipMemsetAsync(s1, 0, (size_t)NN * EMB * 4, stream);
  k_rel_gemm<32><<<NRELS * SPLIT, TPB, 0, stream>>>(h0, W1, perm, offs,
                                                    edge_src, edge_dst, s1);
  k_mean_relu<<<(NN * EMB + TPB - 1) / TPB, TPB, 0, stream>>>(s1, deg, h1);

  // layer 2
  hipMemsetAsync(s2, 0, (size_t)NN * NTYPES * 4, stream);
  k_rel_gemm<16><<<NRELS * SPLIT, TPB, 0, stream>>>(h1, W2, perm, offs,
                                                    edge_src, edge_dst, s2);
  k_mean_out<<<(NN * NTYPES + TPB - 1) / TPB, TPB, 0, stream>>>(s2, deg, y);

  // ponder head
  k_ponder<<<(NN + TPB - 1) / TPB, TPB, 0, stream>>>(y, lw, lb, p);
}
